// DictionaryExpertLISTA_58377195487795
// MI455X (gfx1250) — matrix operand order repaired and checked by reading, not compiled, not run on hardware
//
#include <hip/hip_runtime.h>

#define NLAYERS   5
#define INPUT_DIM 512
#define CODE_DIM  2048
#define BATCH     32768
#define KSEL      32

typedef __attribute__((ext_vector_type(16))) __bf16 v16bf;
typedef __attribute__((ext_vector_type(8)))  float  v8f;

union FragAB { uint4 u4[2]; v16bf v; };
union AccT   { v8f v; float f[8]; };

__device__ __forceinline__ unsigned short f2bf(float f) {
  unsigned u = __float_as_uint(f);
  u += 0x7FFFu + ((u >> 16) & 1u);   // round to nearest even
  return (unsigned short)(u >> 16);
}

// CDNA5 async global->LDS copy (ASYNCcnt-tracked, no VGPR staging).
// %0 = per-lane LDS byte address (low 32 bits of generic pointer), %1 = global addr.
__device__ __forceinline__ void async_fill_b128(const unsigned short* lds_ptr,
                                                const unsigned short* gptr) {
  unsigned lds = (unsigned)(size_t)lds_ptr;
  asm volatile("global_load_async_to_lds_b128 %0, %1, off"
               :: "v"(lds), "v"(gptr)
               : "memory");
}

__device__ __forceinline__ void wait_async0() {
  asm volatile("s_wait_asynccnt 0" ::: "memory");
}

// ---------------- conversion kernels ----------------
__global__ void cast_bf16_kernel(const float* __restrict__ in,
                                 unsigned short* __restrict__ out,
                                 long long total) {
  long long i = (long long)blockIdx.x * blockDim.x + threadIdx.x;
  if (i < total) out[i] = f2bf(in[i]);
}

// out[(l*Nd + n)*Kd + k] = bf16(in[(l*Kd + k)*Nd + n])   (per-layer transpose)
__global__ void cast_transpose_bf16_kernel(const float* __restrict__ in,
                                           unsigned short* __restrict__ out,
                                           int Kd, int Nd, long long total) {
  long long i = (long long)blockIdx.x * blockDim.x + threadIdx.x;
  if (i >= total) return;
  int k = (int)(i % Kd);
  long long rest = i / Kd;
  int n = (int)(rest % Nd);
  long long l = rest / Nd;
  out[i] = f2bf(in[(l * (long long)Kd + k) * Nd + n]);
}

// ---------------- bf16 WMMA GEMM ----------------
// C[M,N] = A[M,K1+K2] * B[K1+K2,N], A given as two row-major bf16 segments
// (A1 stride K1, A2 stride K2), B given as B^T row-major (rows of length K).
// 128x128 block tile, 2x2 waves, 64x64 per wave (16 WMMA / K-step / wave),
// double-buffered LDS filled with async global->LDS.
#define BM  128
#define BN  128
#define BKK 32
#define LDT 40   // padded LDS row stride (ushorts); 40*2=80B, 16B aligned

__global__ __launch_bounds__(128) void gemm_bf16_wmma(
    const unsigned short* __restrict__ A1, int K1,
    const unsigned short* __restrict__ A2, int K2,
    const unsigned short* __restrict__ BT1,
    const unsigned short* __restrict__ BT2,
    float* __restrict__ C, int N) {
  __shared__ __align__(16) unsigned short Al[2][BM * LDT];
  __shared__ __align__(16) unsigned short Bl[2][BN * LDT];

  const int tid  = threadIdx.x;
  const int lane = tid & 31;
  const int w    = tid >> 5;
  const int wM   = w >> 1, wN = w & 1;     // 2 x 2 wave grid, 64x64 each
  const int bN   = blockIdx.x, bM = blockIdx.y;

  AccT acc[4][4];
  #pragma unroll
  for (int i = 0; i < 4; ++i)
    #pragma unroll
    for (int j = 0; j < 4; ++j)
      #pragma unroll
      for (int v = 0; v < 8; ++v) acc[i][j].f[v] = 0.0f;

  const int KT1 = K1 / BKK;
  const int KT  = KT1 + K2 / BKK;

  // issue async fills of one 128x32 A tile + one 128x32 B^T tile (8 per thread)
  auto issue = [&](int kt, int buf) {
    const unsigned short* A;
    const unsigned short* BT;
    int ld, kb;
    if (kt < KT1) { A = A1; BT = BT1; ld = K1; kb = kt * BKK; }
    else          { A = A2; BT = BT2; ld = K2; kb = (kt - KT1) * BKK; }
    #pragma unroll
    for (int s = 0; s < 4; ++s) {
      int c  = tid + s * 128;      // 512 chunks of 8 bf16 per tile
      int r  = c >> 2;
      int k8 = (c & 3) * 8;
      async_fill_b128(&Al[buf][r * LDT + k8],
                      A + ((size_t)(bM * BM + r) * ld + kb + k8));
      async_fill_b128(&Bl[buf][r * LDT + k8],
                      BT + ((size_t)(bN * BN + r) * ld + kb + k8));
    }
  };

  issue(0, 0);
  for (int kt = 0; kt < KT; ++kt) {
    const int cur = kt & 1;
    wait_async0();               // this wave's fills of buf[cur] landed
    __syncthreads();             // everyone's fills of buf[cur] visible
    if (kt + 1 < KT) issue(kt + 1, cur ^ 1);   // overlap fetch with compute

    // B fragments: lane holds column n = lane%16, K run (lane/16)*16..+15
    FragAB fb[4];
    #pragma unroll
    for (int j = 0; j < 4; ++j) {
      int n   = wN * 64 + j * 16 + (lane & 15);
      int off = n * LDT + (lane >> 4) * 16;
      fb[j].u4[0] = *(const uint4*)(&Bl[cur][off]);
      fb[j].u4[1] = *(const uint4*)(&Bl[cur][off + 8]);
    }
    // A fragments: lane holds row m = lane%16, K run (lane/16)*16..+15
    #pragma unroll
    for (int i = 0; i < 4; ++i) {
      FragAB fa;
      int r   = wM * 64 + i * 16 + (lane & 15);
      int off = r * LDT + (lane >> 4) * 16;
      fa.u4[0] = *(const uint4*)(&Al[cur][off]);
      fa.u4[1] = *(const uint4*)(&Al[cur][off + 8]);
      #pragma unroll
      for (int j = 0; j < 4; ++j)
        acc[i][j].v = __builtin_amdgcn_wmma_f32_16x16x32_bf16(
            false, fa.v, false, fb[j].v, (short)0, acc[i][j].v, false, false);
    }
    __syncthreads();             // all waves done reading buf[cur]
  }

  // epilogue: VGPR v -> M = v + (lane/16)*8, N = lane%16
  #pragma unroll
  for (int i = 0; i < 4; ++i) {
    int gM0 = bM * BM + wM * 64 + i * 16 + (lane >> 4) * 8;
    #pragma unroll
    for (int j = 0; j < 4; ++j) {
      int gN = bN * BN + wN * 64 + j * 16 + (lane & 15);
      #pragma unroll
      for (int v = 0; v < 8; ++v)
        C[(size_t)(gM0 + v) * N + gN] = acc[i][j].f[v];
    }
  }
}

// ---------------- exact top-K radix select, one wave per row ----------------
__global__ __launch_bounds__(256) void topk_kernel(
    const float* __restrict__ U, float* __restrict__ Z,
    unsigned short* __restrict__ Zb) {
  __shared__ int hist[8][256];
  __shared__ int bc[8][2];

  const int tid  = threadIdx.x;
  const int lane = tid & 31;
  const int w    = tid >> 5;
  const size_t row = (size_t)blockIdx.x * 8 + w;
  const float* u = U + row * CODE_DIM;

  float    uv[64];
  unsigned key[64];
  #pragma unroll
  for (int i = 0; i < 64; ++i) {               // element index = i*32 + lane
    uv[i]  = u[i * 32 + lane];
    key[i] = __float_as_uint(uv[i]) & 0x7FFFFFFFu;   // |u| bits: monotonic key
  }

  unsigned prefix = 0, pmask = 0;
  int remaining = KSEL;
  #pragma unroll
  for (int level = 0; level < 4; ++level) {
    const int shift = 24 - 8 * level;
    #pragma unroll
    for (int b = 0; b < 8; ++b) hist[w][lane * 8 + b] = 0;
    __syncthreads();
    #pragma unroll
    for (int i = 0; i < 64; ++i)
      if ((key[i] & pmask) == prefix)
        atomicAdd(&hist[w][(key[i] >> shift) & 255], 1);
    __syncthreads();
    if (lane == 0) {
      int cnt = 0, t = 255;
      while (t > 0) {
        int h = hist[w][t];
        if (cnt + h >= remaining) break;
        cnt += h; --t;
      }
      bc[w][0] = t;
      bc[w][1] = remaining - cnt;
    }
    __syncthreads();
    int t = bc[w][0];
    remaining = bc[w][1];
    prefix |= (unsigned)t << shift;
    pmask  |= 0xFFu << shift;
  }

  // keys > prefix are selected; among keys == prefix take `remaining` smallest
  // global indices (chunk-major order == index order) -> deterministic.
  const unsigned thr = prefix;
  int need = remaining;
  #pragma unroll
  for (int i = 0; i < 64; ++i) {
    bool sel = key[i] > thr;
    bool eq  = (key[i] == thr);
    unsigned m = __builtin_amdgcn_ballot_w32(eq);
    if (eq) {
      int rank = __builtin_popcount(m & ((1u << lane) - 1u));
      if (rank < need) sel = true;
    }
    need -= __builtin_popcount(m);
    if (need < 0) need = 0;
    float zv = sel ? uv[i] : 0.0f;
    Z [row * CODE_DIM + i * 32 + lane] = zv;
    Zb[row * CODE_DIM + i * 32 + lane] = f2bf(zv);
  }
}

// ---------------- host orchestration ----------------
extern "C" void kernel_launch(void* const* d_in, const int* in_sizes, int n_in,
                              void* d_out, int out_size, void* d_ws, size_t ws_size,
                              hipStream_t stream) {
  const float* x = (const float*)d_in[0];   // [B, 512]
  const float* W = (const float*)d_in[1];   // [5, 512, 2048]
  const float* S = (const float*)d_in[2];   // [5, 2048, 2048]
  const float* D = (const float*)d_in[3];   // [512, 2048]

  float* recon = (float*)d_out;                         // [B, 512]
  float* z     = recon + (size_t)BATCH * INPUT_DIM;     // [B, 2048]

  // workspace carve-out (256B aligned)
  char* ws = (char*)d_ws;
  size_t off = 0;
  auto carve = [&](size_t bytes) {
    char* p = ws + off;
    off = (off + bytes + 255) & ~(size_t)255;
    return p;
  };
  unsigned short* xb  = (unsigned short*)carve((size_t)BATCH * INPUT_DIM * 2);
  unsigned short* WbT = (unsigned short*)carve((size_t)NLAYERS * CODE_DIM * INPUT_DIM * 2);
  unsigned short* SbT = (unsigned short*)carve((size_t)NLAYERS * CODE_DIM * CODE_DIM * 2);
  unsigned short* Db  = (unsigned short*)carve((size_t)INPUT_DIM * CODE_DIM * 2);
  unsigned short* zb  = (unsigned short*)carve((size_t)BATCH * CODE_DIM * 2);
  float*          u   = (float*)carve((size_t)BATCH * CODE_DIM * 4);

  // one-time conversions
  {
    long long t = (long long)BATCH * INPUT_DIM;
    cast_bf16_kernel<<<(unsigned)(t / 256), 256, 0, stream>>>(x, xb, t);
  }
  {
    long long t = (long long)NLAYERS * CODE_DIM * INPUT_DIM;
    cast_transpose_bf16_kernel<<<(unsigned)(t / 256), 256, 0, stream>>>(
        W, WbT, INPUT_DIM, CODE_DIM, t);
  }
  {
    long long t = (long long)NLAYERS * CODE_DIM * CODE_DIM;
    cast_transpose_bf16_kernel<<<(unsigned)(t / 256), 256, 0, stream>>>(
        S, SbT, CODE_DIM, CODE_DIM, t);
  }
  {
    long long t = (long long)INPUT_DIM * CODE_DIM;   // D already [N=512][K=2048]
    cast_bf16_kernel<<<(unsigned)(t / 256), 256, 0, stream>>>(D, Db, t);
  }

  dim3 gU(CODE_DIM / BN, BATCH / BM);     // (16, 256)
  dim3 gT(BATCH / 8);                     // 4096 blocks, 1 wave per row

  // layer 0: u = x @ W[0]
  gemm_bf16_wmma<<<gU, 128, 0, stream>>>(xb, INPUT_DIM, nullptr, 0,
                                         WbT, nullptr, u, CODE_DIM);
  topk_kernel<<<gT, 256, 0, stream>>>(u, z, zb);

  // layers 1..4: u = [x | z] @ [W[l]; S[l]]  (fused K = 512 + 2048)
  for (int l = 1; l < NLAYERS; ++l) {
    gemm_bf16_wmma<<<gU, 128, 0, stream>>>(
        xb, INPUT_DIM, zb, CODE_DIM,
        WbT + (size_t)l * CODE_DIM * INPUT_DIM,
        SbT + (size_t)l * CODE_DIM * CODE_DIM,
        u, CODE_DIM);
    topk_kernel<<<gT, 256, 0, stream>>>(u, z, zb);
  }

  // recon = z @ D^T  (B^T rows = D rows, K = 2048, N = 512)
  dim3 gR(INPUT_DIM / BN, BATCH / BM);    // (4, 256)
  gemm_bf16_wmma<<<gR, 128, 0, stream>>>(zb, CODE_DIM, nullptr, 0,
                                         Db, nullptr, recon, INPUT_DIM);
}
